// MultiHeadAttention_2250562863251
// MI455X (gfx1250) — matrix operand order repaired and checked by reading, not compiled, not run on hardware
//
#include <hip/hip_runtime.h>
#include <hip/hip_bf16.h>

// MultiHeadAttention (rank-1-logit variant) for gfx1250 / MI455X, wave32.
// Heavy stages: v_wmma_f32_16x16x32_bf16 (f32 accumulate).
// Attention V tiles: double-buffered GLOBAL_LOAD_ASYNC_TO_LDS_B128 (ASYNCcnt).
// Softmax statistics (qs, ks, max, sum, exp) stay in f32 for fidelity.

#define NEGV (-1000.0f)

typedef __attribute__((ext_vector_type(16))) __bf16 v16bf;
typedef __attribute__((ext_vector_type(8)))  float  v8f;

union Frag16 {           // one WMMA 16-bit operand fragment (32 B / lane)
  uint4 q[2];
  unsigned short us[16];
  v16bf bf;
};
union F16x { float4 v4[4]; float f[16]; };

__device__ __forceinline__ unsigned short f2bf(float f) {
  unsigned int u = __float_as_uint(f);
  u += 0x7FFFu + ((u >> 16) & 1u);           // round-to-nearest-even
  return (unsigned short)(u >> 16);
}

// Async global->LDS copy, 16 B per lane. LDS byte offset = low 32 bits of the
// generic pointer to __shared__ (AS3->generic addrspacecast keeps the offset
// in addr[31:0]; hardware adds the VDST value to the LDS base, see ISA 15.18.3).
__device__ __forceinline__ void async_load_b128(unsigned lds_off, const void* g) {
  asm volatile("global_load_async_to_lds_b128 %0, %1, off"
               :
               : "v"(lds_off), "v"((unsigned long long)(uintptr_t)g)
               : "memory");
}
__device__ __forceinline__ void wait_async0() {
  asm volatile("s_wait_asynccnt 0" ::: "memory");
}

// ---------------------------------------------------------------------------
// 1) Per-head row-sum reduction of Wq / Wk (+ bias sums).
//    Wr[h][e] = sum_{i in head h} W[h*64+i][e]
// ---------------------------------------------------------------------------
__global__ void wred_kernel(const float* __restrict__ Wq, const float* __restrict__ bq,
                            const float* __restrict__ Wk, const float* __restrict__ bk,
                            float* __restrict__ Wqr, float* __restrict__ bqr,
                            float* __restrict__ Wkr, float* __restrict__ bkr) {
  const int h   = blockIdx.x;      // 0..15
  const int sel = blockIdx.y;      // 0 = q, 1 = k
  const float* W  = sel ? Wk  : Wq;
  const float* bb = sel ? bk  : bq;
  float* Wr = sel ? Wkr : Wqr;
  float* br = sel ? bkr : bqr;
  for (int e = threadIdx.x; e < 1024; e += blockDim.x) {
    float s = 0.f;
#pragma unroll 4
    for (int i = 0; i < 64; ++i) s += W[(h * 64 + i) * 1024 + e];
    Wr[h * 1024 + e] = s;
  }
  if (threadIdx.x == 0) {
    float s = 0.f;
    for (int i = 0; i < 64; ++i) s += bb[h * 64 + i];
    br[h] = s;
  }
}

// ---------------------------------------------------------------------------
// 2) qs[r][h] = X[r,:] . Wr[h,:] + br[h]   (r = m*2 + b, full f32 precision)
// ---------------------------------------------------------------------------
__global__ __launch_bounds__(256) void qsks_kernel(
    const float* __restrict__ queries, const float* __restrict__ keys,
    const float* __restrict__ Wqr, const float* __restrict__ Wkr,
    const float* __restrict__ bqr, const float* __restrict__ bkr,
    float* __restrict__ qs, float* __restrict__ ks) {
  __shared__ __align__(16) float sx[1024];
  const int r   = blockIdx.x;      // 0..4095
  const int sel = blockIdx.y;      // 0 = q, 1 = k
  const float* X  = sel ? keys : queries;
  const float* Wr = sel ? Wkr  : Wqr;
  const float* br = sel ? bkr  : bqr;
  float* out = sel ? ks : qs;
  const int t = threadIdx.x;
  ((float4*)sx)[t] = ((const float4*)(X + (size_t)r * 1024))[t];
  __syncthreads();
  const int h = t >> 4;            // 16 heads x 16 lanes
  const int g = t & 15;
  const float* wrow = Wr + h * 1024;
  float p = 0.f;
#pragma unroll 8
  for (int e = g; e < 1024; e += 16) p += sx[e] * wrow[e];
  p += __shfl_down(p, 8, 16);
  p += __shfl_down(p, 4, 16);
  p += __shfl_down(p, 2, 16);
  p += __shfl_down(p, 1, 16);
  if (g == 0) out[r * 16 + h] = p + br[h];
}

// ---------------------------------------------------------------------------
// 3/5) bf16 WMMA GEMM:  C[4096x1024] = A[4096x1024] * W^T + bias
//      ABF16: 0 = A is f32 (converted while staging), 1 = A is bf16
//      OUTVT: 1 = write bf16 Vt[b][h][d][n]; 0 = write f32 row-major
//      Block = 256 thr (8 waves) -> 128x128 C tile, K staged 32 at a time.
// ---------------------------------------------------------------------------
template <int ABF16, int OUTVT>
__global__ __launch_bounds__(256) void gemm_kernel(const void* __restrict__ Ap,
                                                   const float* __restrict__ W,
                                                   const float* __restrict__ bias,
                                                   void* __restrict__ Outp) {
  __shared__ __align__(16) unsigned short sA[128 * 40];  // pad 32->40 shorts/row
  __shared__ __align__(16) unsigned short sB[128 * 40];
  const int t   = threadIdx.x;
  const int wv  = t >> 5;
  const int L   = t & 31;
  const int l16 = L & 15;
  const int hf  = (L >> 4) & 1;
  const int rows0 = blockIdx.y * 128;
  const int cols0 = blockIdx.x * 128;

  v8f acc[8] = {};

  for (int k0 = 0; k0 < 1024; k0 += 32) {
    __syncthreads();
    if (ABF16) {
      const unsigned short* Abf = (const unsigned short*)Ap;
      const int row = t >> 2, c = t & 3;
#pragma unroll
      for (int p = 0; p < 2; ++p) {
        const int r = row + p * 64;
        uint4 v = *(const uint4*)(Abf + (size_t)(rows0 + r) * 1024 + k0 + c * 8);
        *(uint4*)(sA + r * 40 + c * 8) = v;
      }
      if (k0 + 32 < 1024)
        __builtin_prefetch(Abf + (size_t)(rows0 + row) * 1024 + k0 + 32 + c * 8, 0, 1);
    } else {
      const float* Af = (const float*)Ap;
      const int row = t >> 3, c = t & 7;
#pragma unroll
      for (int p = 0; p < 4; ++p) {
        const int r = row + p * 32;
        float4 v = *(const float4*)(Af + (size_t)(rows0 + r) * 1024 + k0 + c * 4);
        uint2 pk;
        pk.x = (unsigned)f2bf(v.x) | ((unsigned)f2bf(v.y) << 16);
        pk.y = (unsigned)f2bf(v.z) | ((unsigned)f2bf(v.w) << 16);
        *(uint2*)(sA + r * 40 + c * 4) = pk;
      }
      if (k0 + 32 < 1024)
        __builtin_prefetch(Af + (size_t)(rows0 + row) * 1024 + k0 + 32 + c * 4, 0, 1);
    }
    {  // B tile: B[k][j] = W[j][k]; stage W rows cols0..cols0+127 as bf16
      const int row = t >> 3, c = t & 7;
#pragma unroll
      for (int p = 0; p < 4; ++p) {
        const int r = row + p * 32;
        float4 v = *(const float4*)(W + (size_t)(cols0 + r) * 1024 + k0 + c * 4);
        uint2 pk;
        pk.x = (unsigned)f2bf(v.x) | ((unsigned)f2bf(v.y) << 16);
        pk.y = (unsigned)f2bf(v.z) | ((unsigned)f2bf(v.w) << 16);
        *(uint2*)(sB + r * 40 + c * 4) = pk;
      }
      if (k0 + 32 < 1024)
        __builtin_prefetch(W + (size_t)(cols0 + row) * 1024 + k0 + 32 + c * 4, 0, 1);
    }
    __syncthreads();

    // A fragment: lane row = wv*16 + l16; k-runs at half*8 and 16+half*8
    Frag16 a;
    const unsigned short* ab = sA + (wv * 16 + l16) * 40 + hf * 8;
    a.q[0] = *(const uint4*)(ab);
    a.q[1] = *(const uint4*)(ab + 16);
#pragma unroll
    for (int f = 0; f < 8; ++f) {  // B fragment: lane col; k-runs at half*8 and 16+half*8
      Frag16 bf;
      const unsigned short* bb = sB + (f * 16 + l16) * 40 + hf * 8;
      bf.q[0] = *(const uint4*)(bb);
      bf.q[1] = *(const uint4*)(bb + 16);
      acc[f] = __builtin_amdgcn_wmma_f32_16x16x32_bf16(false, a.bf, false, bf.bf,
                                                       (short)0, acc[f], false, false);
    }
  }

#pragma unroll
  for (int f = 0; f < 8; ++f) {
    const int j  = cols0 + f * 16 + l16;
    const float bj = bias[j];
#pragma unroll
    for (int r = 0; r < 8; ++r) {
      const int R = rows0 + wv * 16 + r + 8 * hf;  // C row: m = r + 8*half
      const float val = acc[f][r] + bj;
      if (OUTVT) {  // Vt[b][h][d][n] bf16 (n innermost, contiguous for attn B)
        unsigned short* Vt = (unsigned short*)Outp;
        const int nseq = R >> 1, bidx = R & 1;
        const int hh = j >> 6, dd = j & 63;
        Vt[(size_t)((bidx * 16 + hh) * 64 + dd) * 2048 + nseq] = f2bf(val);
      } else {
        ((float*)Outp)[(size_t)R * 1024 + j] = val;
      }
    }
  }
}

// ---------------------------------------------------------------------------
// 4) Flash-style streaming softmax over rank-1 logits + WMMA P*V.
//    grid = (32 bh, 16 m-groups); block = 8 waves; wave owns 16 m-rows.
//    V tiles double-buffered in LDS via async global->LDS DMA (ASYNCcnt).
// ---------------------------------------------------------------------------
__global__ __launch_bounds__(256) void attn_kernel(
    const float* __restrict__ qs, const float* __restrict__ ks,
    const unsigned short* __restrict__ Vt, const unsigned char* __restrict__ mask,
    unsigned short* __restrict__ pooled) {
  __shared__ __align__(16) unsigned short sV[2][64 * 40];  // [d][k] bf16, padded
  __shared__ __align__(16) float sKA[2][32];
  __shared__ __align__(16) float sKB[2][32];
  const int t   = threadIdx.x;
  const int wv  = t >> 5;
  const int L   = t & 31;
  const int l16 = L & 15;
  const int hf  = (L >> 4) & 1;
  const int bh  = blockIdx.x;
  const int b   = bh & 1;
  const int h   = bh >> 1;
  const int m0  = blockIdx.y * 128 + wv * 16;
  const int m_glob = m0 + l16;

  const size_t vbase = (size_t)((b * 16 + h) * 64) * 2048;
  const float qsv = qs[(size_t)(m_glob * 2 + b) * 16 + h];

  v8f acc[4] = {};
  float runmax = -INFINITY;
  float runsum = 0.f;

  const int vd = t >> 2;  // 0..63 staging d-row, 16 B (8 bf16) per thread
  const int vc = t & 3;
  const unsigned short* gV = Vt + vbase + (size_t)vd * 2048 + vc * 8;
  const unsigned ldsV[2] = {
      (unsigned)(uintptr_t)(&sV[0][vd * 40 + vc * 8]),
      (unsigned)(uintptr_t)(&sV[1][vd * 40 + vc * 8])};

  // prologue: chunk 0 in flight
  async_load_b128(ldsV[0], gV);
  if (t < 32) {  // bake padding mask into (ksA, ksB): l = qs*ksA + ksB
    const float kv = ks[(size_t)(t * 2 + b) * 16 + h];
    const bool mk = mask[t * 2 + b] != 0;
    sKA[0][t] = mk ? 0.f : kv;
    sKB[0][t] = mk ? NEGV : 0.f;
  }

  int cur = 0;
  for (int n0 = 0; n0 < 2048; n0 += 32) {
    wait_async0();       // my async V loads for this chunk have landed in LDS
    __syncthreads();     // everyone's V + ks staged; prior chunk's reads done

    const int nxt = n0 + 32;
    if (nxt < 2048) {    // overlap next tile DMA with this chunk's compute
      async_load_b128(ldsV[cur ^ 1], gV + nxt);
      if (t < 32) {
        const int n = nxt + t;
        const float kv = ks[(size_t)(n * 2 + b) * 16 + h];
        const bool mk = mask[n * 2 + b] != 0;
        sKA[cur ^ 1][t] = mk ? 0.f : kv;
        sKB[cur ^ 1][t] = mk ? NEGV : 0.f;
      }
    }

    F16x KA, KB;  // per-lane k-runs matching the A fragment element order
    {
      const float4* pa = (const float4*)sKA[cur];
      const float4* pb = (const float4*)sKB[cur];
      KA.v4[0] = pa[hf * 2];     KA.v4[1] = pa[hf * 2 + 1];
      KA.v4[2] = pa[4 + hf * 2]; KA.v4[3] = pa[5 + hf * 2];
      KB.v4[0] = pb[hf * 2];     KB.v4[1] = pb[hf * 2 + 1];
      KB.v4[2] = pb[4 + hf * 2]; KB.v4[3] = pb[5 + hf * 2];
    }
    float l[16];
    float tmax = -INFINITY;
#pragma unroll
    for (int e = 0; e < 16; ++e) {
      const int k = (e < 8) ? (hf * 8 + e) : (16 + hf * 8 + (e - 8));
      float lv = fmaf(qsv, KA.f[e], KB.f[e]);
      lv = (n0 + k < m_glob) ? NEGV : lv;  // faithful inverted-causal mask
      l[e] = lv;
      tmax = fmaxf(tmax, lv);
    }
    tmax = fmaxf(tmax, __shfl_xor(tmax, 16, 32));   // combine row halves
    const float nmax  = fmaxf(runmax, tmax);
    const float scale = __expf(runmax - nmax);      // exp(-inf)=0 first iter
    runmax = nmax;

    Frag16 aP;
    float lsum = 0.f;
#pragma unroll
    for (int e = 0; e < 16; ++e) {
      const float pe = __expf(l[e] - nmax);
      lsum += pe;
      aP.us[e] = f2bf(pe);
    }
    runsum = runsum * scale + lsum;

    v8f sclv;  // C rows are r + 8*half -> fetch that row's scale via shuffle
#pragma unroll
    for (int r = 0; r < 8; ++r) sclv[r] = __shfl(scale, r + 8 * hf, 32);

#pragma unroll
    for (int f = 0; f < 4; ++f) {  // d-chunks 0,16,32,48
      Frag16 bV;
      const unsigned short* vb = &sV[cur][(f * 16 + l16) * 40 + hf * 8];
      bV.q[0] = *(const uint4*)(vb);
      bV.q[1] = *(const uint4*)(vb + 16);
      acc[f] = acc[f] * sclv;
      acc[f] = __builtin_amdgcn_wmma_f32_16x16x32_bf16(false, aP.bf, false, bV.bf,
                                                       (short)0, acc[f], false, false);
    }
    cur ^= 1;
  }

  const float tot = runsum + __shfl_xor(runsum, 16, 32);
  const float inv = 1.0f / tot;   // >= 1 term of exp(0) always present
  v8f invv;
#pragma unroll
  for (int r = 0; r < 8; ++r) invv[r] = __shfl(inv, r + 8 * hf, 32);
#pragma unroll
  for (int f = 0; f < 4; ++f)
#pragma unroll
    for (int r = 0; r < 8; ++r) {
      const int mg = m0 + r + 8 * hf;
      pooled[(size_t)(mg * 2 + b) * 1024 + h * 64 + f * 16 + l16] =
          f2bf(acc[f][r] * invv[r]);
    }
}

// ---------------------------------------------------------------------------
extern "C" void kernel_launch(void* const* d_in, const int* in_sizes, int n_in,
                              void* d_out, int out_size, void* d_ws, size_t ws_size,
                              hipStream_t stream) {
  (void)in_sizes; (void)n_in; (void)out_size; (void)ws_size;
  const float* queries = (const float*)d_in[0];
  const float* keys    = (const float*)d_in[1];
  const float* values  = (const float*)d_in[2];
  const float* Wq = (const float*)d_in[3];
  const float* bq = (const float*)d_in[4];
  const float* Wk = (const float*)d_in[5];
  const float* bk = (const float*)d_in[6];
  const float* Wv = (const float*)d_in[7];
  const float* bv = (const float*)d_in[8];
  const float* Wo = (const float*)d_in[9];
  const float* bo = (const float*)d_in[10];
  const unsigned char* in_mask = (const unsigned char*)d_in[11];  // jnp bool_
  float* out = (float*)d_out;

  char* p = (char*)d_ws;                               // ~17.4 MB scratch
  float* Wqr = (float*)p;          p += 16 * 1024 * 4;
  float* Wkr = (float*)p;          p += 16 * 1024 * 4;
  float* bqr = (float*)p;          p += 256;
  float* bkr = (float*)p;          p += 256;
  float* qsb = (float*)p;          p += 4096 * 16 * 4;
  float* ksb = (float*)p;          p += 4096 * 16 * 4;
  unsigned short* Vt     = (unsigned short*)p; p += (size_t)2 * 16 * 64 * 2048 * 2;
  unsigned short* pooled = (unsigned short*)p; p += (size_t)4096 * 1024 * 2;

  wred_kernel<<<dim3(16, 2), 256, 0, stream>>>(Wq, bq, Wk, bk, Wqr, bqr, Wkr, bkr);
  qsks_kernel<<<dim3(4096, 2), 256, 0, stream>>>(queries, keys, Wqr, Wkr, bqr, bkr,
                                                 qsb, ksb);
  gemm_kernel<0, 1><<<dim3(8, 32), 256, 0, stream>>>(values, Wv, bv, Vt);
  attn_kernel<<<dim3(32, 16), 256, 0, stream>>>(qsb, ksb, Vt, in_mask, pooled);
  gemm_kernel<1, 0><<<dim3(8, 32), 256, 0, stream>>>(pooled, Wo, bo, out);
}
